// DynamicTemporalModel_19335942767173
// MI455X (gfx1250) — matrix operand order repaired and checked by reading, not compiled, not run on hardware
//
#include <hip/hip_runtime.h>

// ---------------------------------------------------------------------------
// DynamicTemporalModel fused implementation for MI455X (gfx1250, wave32).
//
// Math reduction (T=1):
//   attention is identity on v; LSTM f-gate multiplies c0=0 (dead);
//   g = M' @ x[b] + dbias[stage[b]],  M' = Wg'(1536x512) @ Wo @ Wv
//   h = sigmoid(o)*tanh(sigmoid(i)*tanh(g));  logits = head_w @ h + head_b
// ---------------------------------------------------------------------------

typedef __attribute__((ext_vector_type(16))) __bf16 v16bf;
typedef __attribute__((ext_vector_type(8)))  float  v8f;
typedef __attribute__((__vector_size__(16))) int    vi4;   // async builtin pointee

union BFrag { uint4 q[2]; v16bf v; };
union AFrag { unsigned int u[8]; v16bf v; };

#define B_ROWS 65536
#define DMODEL 512
#define NOUT   1536   // i, g, o blocks (f elided)

#if __has_builtin(__builtin_amdgcn_global_load_async_to_lds_b128)
#define HAVE_ASYNC 1
#else
#define HAVE_ASYNC 0
#endif

__device__ __forceinline__ unsigned short f2bf(float f) {   // RNE (weights)
  unsigned int u = __float_as_uint(f);
  unsigned int r = u + 0x7fffu + ((u >> 16) & 1u);
  return (unsigned short)(r >> 16);
}
// Hot-loop activation pack: truncating, 1 v_perm_b32 per pair.
__device__ __forceinline__ unsigned int packbits(unsigned int lo, unsigned int hi) {
#if __has_builtin(__builtin_amdgcn_perm)
  return __builtin_amdgcn_perm(hi, lo, 0x07060302u);  // [lo.b2,lo.b3,hi.b2,hi.b3]
#else
  return (lo >> 16) | (hi & 0xffff0000u);
#endif
}
__device__ __forceinline__ float bf2f(unsigned short h) {
  return __uint_as_float(((unsigned int)h) << 16);
}
__device__ __forceinline__ float fast_rcp(float x) {
#if __has_builtin(__builtin_amdgcn_rcpf)
  return __builtin_amdgcn_rcpf(x);
#else
  return 1.0f / x;
#endif
}
__device__ __forceinline__ float fast_sigm(float x) {
  return fast_rcp(1.0f + __expf(-x));
}
__device__ __forceinline__ float fast_tanh(float x) {
  float ax = fabsf(x);
  float t  = __expf(-2.0f * ax);
  float r  = (1.0f - t) * fast_rcp(1.0f + t);
  return copysignf(r, x);
}

#if HAVE_ASYNC
typedef __attribute__((address_space(1))) vi4 gvi4;
typedef __attribute__((address_space(3))) vi4 lvi4;
__device__ __forceinline__ void async_b128(const void* g, void* l) {
  __builtin_amdgcn_global_load_async_to_lds_b128(
      (gvi4*)(g), (lvi4*)(l), 0, 0);
}
__device__ __forceinline__ void wait_async0() {
#if __has_builtin(__builtin_amdgcn_s_wait_asynccnt)
  __builtin_amdgcn_s_wait_asynccnt(0);
#else
  asm volatile("s_wait_asynccnt 0x0" ::: "memory");
#endif
}
#endif

// --------------------------- precompute kernels ----------------------------

// pos[s][d] = enc_b2[d] + sum_e relu(-s*enc_w1[e]+enc_b1[e]) * enc_w2[d][e]
__global__ void pos_kernel(const float* __restrict__ enc_w1,
                           const float* __restrict__ enc_b1,
                           const float* __restrict__ enc_w2,
                           const float* __restrict__ enc_b2,
                           float* __restrict__ posBuf) {
  int id = blockIdx.x * 256 + threadIdx.x;
  if (id >= 5 * DMODEL) return;
  int s = id >> 9, d = id & 511;
  float acc = enc_b2[d];
  float sf = -(float)s;
  #pragma unroll 8
  for (int e = 0; e < 64; ++e) {
    float h = fmaf(sf, enc_w1[e], enc_b1[e]);
    h = fmaxf(h, 0.0f);
    acc = fmaf(h, enc_w2[d * 64 + e], acc);
  }
  posBuf[s * DMODEL + d] = acc;
}

// Wov[e][d] = sum_k out_proj_w[e][k] * in_proj_w[1024+k][d]   (512x512 f32)
__global__ void wov_kernel(const float* __restrict__ out_proj_w,
                           const float* __restrict__ in_proj_w,
                           float* __restrict__ Wov) {
  int tid = threadIdx.x;
  int e0  = blockIdx.x * 4;
  float a0[4] = {0.f,0.f,0.f,0.f};
  float a1[4] = {0.f,0.f,0.f,0.f};
  for (int k = 0; k < 512; ++k) {
    float v0 = in_proj_w[(size_t)(1024 + k) * 512 + tid];
    float v1 = in_proj_w[(size_t)(1024 + k) * 512 + tid + 256];
    #pragma unroll
    for (int i = 0; i < 4; ++i) {
      float w = out_proj_w[(size_t)(e0 + i) * 512 + k];
      a0[i] = fmaf(w, v0, a0[i]);
      a1[i] = fmaf(w, v1, a1[i]);
    }
  }
  #pragma unroll
  for (int i = 0; i < 4; ++i) {
    Wov[(size_t)(e0 + i) * 512 + tid]       = a0[i];
    Wov[(size_t)(e0 + i) * 512 + tid + 256] = a1[i];
  }
}

// bao[e] = out_proj_b[e] + sum_k out_proj_w[e][k] * in_proj_b[1024+k]
__global__ void bao_kernel(const float* __restrict__ out_proj_w,
                           const float* __restrict__ out_proj_b,
                           const float* __restrict__ in_proj_b,
                           float* __restrict__ bao) {
  int e = blockIdx.x * 256 + threadIdx.x;
  if (e >= 512) return;
  float acc = out_proj_b[e];
  for (int k = 0; k < 512; ++k)
    acc = fmaf(out_proj_w[(size_t)e * 512 + k], in_proj_b[1024 + k], acc);
  bao[e] = acc;
}

// Mrow[n][k] (bf16) = sum_j lstm_w_ih[r(n)][j] * Wov[j][k], r skips f-gate rows
__global__ void mrow_kernel(const float* __restrict__ lstm_w_ih,
                            const float* __restrict__ Wov,
                            unsigned short* __restrict__ Mrow) {
  int tid = threadIdx.x;
  int n0  = blockIdx.x * 4;
  float a0[4] = {0.f,0.f,0.f,0.f};
  float a1[4] = {0.f,0.f,0.f,0.f};
  for (int j = 0; j < 512; ++j) {
    float v0 = Wov[(size_t)j * 512 + tid];
    float v1 = Wov[(size_t)j * 512 + tid + 256];
    #pragma unroll
    for (int i = 0; i < 4; ++i) {
      int n = n0 + i;
      int r = n + (n >= 512 ? 512 : 0);
      float w = lstm_w_ih[(size_t)r * 512 + j];
      a0[i] = fmaf(w, v0, a0[i]);
      a1[i] = fmaf(w, v1, a1[i]);
    }
  }
  #pragma unroll
  for (int i = 0; i < 4; ++i) {
    Mrow[(size_t)(n0 + i) * 512 + tid]       = f2bf(a0[i]);
    Mrow[(size_t)(n0 + i) * 512 + tid + 256] = f2bf(a1[i]);
  }
}

// dbias[st][n] = lstm_b_ih[r]+lstm_b_hh[r] + Wg[r]·bao + M'[n]·pos[st]
__global__ void dbias_kernel(const unsigned short* __restrict__ Mrow,
                             const float* __restrict__ posBuf,
                             const float* __restrict__ lstm_w_ih,
                             const float* __restrict__ bao,
                             const float* __restrict__ lstm_b_ih,
                             const float* __restrict__ lstm_b_hh,
                             float* __restrict__ dbias) {
  int id = blockIdx.x * 256 + threadIdx.x;
  if (id >= 5 * NOUT) return;
  int st = id / NOUT, n = id % NOUT;
  int r = n + (n >= 512 ? 512 : 0);
  float acc = lstm_b_ih[r] + lstm_b_hh[r];
  for (int k = 0; k < 512; ++k)
    acc = fmaf(lstm_w_ih[(size_t)r * 512 + k], bao[k], acc);
  for (int k = 0; k < 512; ++k)
    acc = fmaf(bf2f(Mrow[(size_t)n * 512 + k]), posBuf[st * 512 + k], acc);
  dbias[(size_t)st * NOUT + n] = acc;
}

// logits init: out_logits[b][t] = head_b[t]; main kernel atomically adds partials
__global__ void init_logits_kernel(const float* __restrict__ head_b,
                                   float* __restrict__ out_logits) {
  int id = blockIdx.x * 256 + threadIdx.x;
  if (id < B_ROWS * 5) out_logits[id] = head_b[id % 5];
}

// ------------------------------ main kernel --------------------------------
// grid(8, 512): x = col chunk c (64 h-cols), y = row tile (128 batch rows).
// Each wave: 16 rows, 3x(16x64) f32 accumulators (i,g,o), K-loop 512/32.
// A tile kept in f32 in LDS (async-to-LDS b128 copies); bf16 pack at
// fragment-load time (v_perm_b32). B tile is bf16 fused weights.

__global__ __launch_bounds__(256) void fused_main_kernel(
    const float* __restrict__ x, const int* __restrict__ stage,
    const unsigned short* __restrict__ Mrow, const float* __restrict__ dbias,
    const float* __restrict__ head_w,
    float* __restrict__ out_feat, float* __restrict__ out_logits) {

  __shared__ __align__(16) float          Ashf[2][128][32];    // 32 KB (f32)
  __shared__ __align__(16) unsigned short Bsh[2][3][64][32];   // 24 KB (bf16)
  __shared__ int   Sstage[128];
  __shared__ float Sdbias[3][5][64];
  __shared__ float Shw[5][64];

  const int tid  = threadIdx.x;
  const int wave = tid >> 5;
  const int lane = tid & 31;
  const int colBase = blockIdx.x * 64;     // within the 512 h-columns
  const int rowBase = blockIdx.y * 128;

  for (int i = tid; i < 128; i += 256) Sstage[i] = stage[rowBase + i];
  for (int i = tid; i < 3 * 5 * 64; i += 256) {
    int s = i / 320, r = i % 320, st = r >> 6, n = r & 63;
    Sdbias[s][st][n] = dbias[(size_t)st * NOUT + s * 512 + colBase + n];
  }
  for (int i = tid; i < 5 * 64; i += 256)
    Shw[i >> 6][i & 63] = head_w[(size_t)(i >> 6) * 512 + colBase + (i & 63)];

  const v8f vzero = {0.f,0.f,0.f,0.f,0.f,0.f,0.f,0.f};
  v8f acc[3][4];
  #pragma unroll
  for (int s = 0; s < 3; ++s)
    #pragma unroll
    for (int j = 0; j < 4; ++j) acc[s][j] = vzero;

  // fragment compute on one LDS buffer: 12 WMMAs
  auto compute_tile = [&](int buf) {
    const int rloc = wave * 16 + (lane & 15);
    const int kbA  = (lane < 16) ? 0 : 8;    // A: K {kb..kb+7, kb+16..kb+23}
    uint4 ua0 = *(const uint4*)&Ashf[buf][rloc][kbA];
    uint4 ua1 = *(const uint4*)&Ashf[buf][rloc][kbA + 4];
    uint4 ua2 = *(const uint4*)&Ashf[buf][rloc][kbA + 16];
    uint4 ua3 = *(const uint4*)&Ashf[buf][rloc][kbA + 20];
    AFrag a;
    a.u[0] = packbits(ua0.x, ua0.y); a.u[1] = packbits(ua0.z, ua0.w);
    a.u[2] = packbits(ua1.x, ua1.y); a.u[3] = packbits(ua1.z, ua1.w);
    a.u[4] = packbits(ua2.x, ua2.y); a.u[5] = packbits(ua2.z, ua2.w);
    a.u[6] = packbits(ua3.x, ua3.y); a.u[7] = packbits(ua3.z, ua3.w);
    const int kbB = (lane < 16) ? 0 : 8;     // B: K {kb..kb+7, kb+16..kb+23}
    #pragma unroll
    for (int s = 0; s < 3; ++s) {
      #pragma unroll
      for (int j = 0; j < 4; ++j) {
        int nloc = j * 16 + (lane & 15);
        BFrag b;
        b.q[0] = *(const uint4*)&Bsh[buf][s][nloc][kbB];
        b.q[1] = *(const uint4*)&Bsh[buf][s][nloc][kbB + 16];
        acc[s][j] = __builtin_amdgcn_wmma_f32_16x16x32_bf16(
            false, a.v, false, b.v, (short)0, acc[s][j], false, false);
      }
    }
  };

#if HAVE_ASYNC
  // ---- async global->LDS pipeline (ASYNCcnt) ----
  auto issue_tile = [&](int b, int k0) {
    #pragma unroll
    for (int i = 0; i < 4; ++i) {           // A: 128x32 f32 = 1024 x 16B
      int idx = i * 256 + tid;
      int row = idx >> 3, kc = (idx & 7) << 2;
      async_b128(&x[(size_t)(rowBase + row) * DMODEL + k0 + kc],
                 &Ashf[b][row][kc]);
    }
    #pragma unroll
    for (int i = 0; i < 3; ++i) {           // B: 3x64x32 bf16 = 768 x 16B
      int idx = i * 256 + tid;
      int s = idx >> 8, r = idx & 255, nloc = r >> 2, k8 = (r & 3) << 3;
      async_b128(&Mrow[(size_t)(s * 512 + colBase + nloc) * DMODEL + k0 + k8],
                 &Bsh[b][s][nloc][k8]);
    }
  };

  issue_tile(0, 0);
  wait_async0();
  __syncthreads();
  #pragma unroll
  for (int kt = 0; kt < 16; ++kt) {
    const int buf = kt & 1;
    if (kt < 15) issue_tile(buf ^ 1, (kt + 1) * 32);  // overlap with compute
    compute_tile(buf);
    wait_async0();
    __syncthreads();
  }
#else
  // ---- fallback: global->reg prefetch, reg->LDS commit ----
  #pragma unroll
  for (int i = 0; i < 4; ++i) {
    int idx = i * 256 + tid;
    int row = idx >> 3, kc = (idx & 7) << 2;
    *(uint4*)&Ashf[0][row][kc] =
        *(const uint4*)&x[(size_t)(rowBase + row) * DMODEL + kc];
  }
  #pragma unroll
  for (int i = 0; i < 3; ++i) {
    int idx = i * 256 + tid;
    int s = idx >> 8, r = idx & 255, nloc = r >> 2, k8 = (r & 3) << 3;
    *(uint4*)&Bsh[0][s][nloc][k8] =
        *(const uint4*)&Mrow[(size_t)(s * 512 + colBase + nloc) * DMODEL + k8];
  }
  __syncthreads();

  #pragma unroll
  for (int kt = 0; kt < 16; ++kt) {
    const int buf = kt & 1;
    uint4 aregq[4];
    uint4 breg[3];
    const int k0n = (kt + 1) * 32;
    if (kt < 15) {
      #pragma unroll
      for (int i = 0; i < 4; ++i) {
        int idx = i * 256 + tid;
        int row = idx >> 3, kc = (idx & 7) << 2;
        aregq[i] = *(const uint4*)&x[(size_t)(rowBase + row) * DMODEL + k0n + kc];
      }
      #pragma unroll
      for (int i = 0; i < 3; ++i) {
        int idx = i * 256 + tid;
        int s = idx >> 8, r = idx & 255, nloc = r >> 2, k8 = (r & 3) << 3;
        breg[i] = *(const uint4*)&Mrow[(size_t)(s * 512 + colBase + nloc) * DMODEL + k0n + k8];
      }
    }
    compute_tile(buf);
    if (kt < 15) {
      #pragma unroll
      for (int i = 0; i < 4; ++i) {
        int idx = i * 256 + tid;
        int row = idx >> 3, kc = (idx & 7) << 2;
        *(uint4*)&Ashf[buf ^ 1][row][kc] = aregq[i];
      }
      #pragma unroll
      for (int i = 0; i < 3; ++i) {
        int idx = i * 256 + tid;
        int s = idx >> 8, r = idx & 255, nloc = r >> 2, k8 = (r & 3) << 3;
        *(uint4*)&Bsh[buf ^ 1][s][nloc][k8] = breg[i];
      }
    }
    __syncthreads();
  }
#endif

  // ---- epilogue: bias + LSTM nonlinearity + store h + logits partials ----
  const int m0      = (lane < 16) ? 0 : 8;   // C/D: VGPR j holds rows j / j+8
  const int colLane = lane & 15;
  #pragma unroll
  for (int vg = 0; vg < 8; ++vg) {
    const int mloc = wave * 16 + m0 + vg;
    const int row  = rowBase + mloc;
    const int st   = Sstage[mloc];
    float lsum[5] = {0.f,0.f,0.f,0.f,0.f};
    #pragma unroll
    for (int j = 0; j < 4; ++j) {
      const int n = j * 16 + colLane;
      float iv = acc[0][j][vg] + Sdbias[0][st][n];
      float gv = acc[1][j][vg] + Sdbias[1][st][n];
      float ov = acc[2][j][vg] + Sdbias[2][st][n];
      float ct = fast_sigm(iv) * fast_tanh(gv);
      float h  = fast_sigm(ov) * fast_tanh(ct);
      out_feat[(size_t)row * DMODEL + colBase + n] = h;
      #pragma unroll
      for (int t = 0; t < 5; ++t) lsum[t] = fmaf(h, Shw[t][n], lsum[t]);
    }
    #pragma unroll
    for (int t = 0; t < 5; ++t) {
      float v = lsum[t];
      v += __shfl_xor(v, 1, 32);
      v += __shfl_xor(v, 2, 32);
      v += __shfl_xor(v, 4, 32);
      v += __shfl_xor(v, 8, 32);
      if (colLane == 0) atomicAdd(&out_logits[(size_t)row * 5 + t], v);
    }
  }
}

// ------------------------------- launch ------------------------------------

extern "C" void kernel_launch(void* const* d_in, const int* in_sizes, int n_in,
                              void* d_out, int out_size, void* d_ws, size_t ws_size,
                              hipStream_t stream) {
  const float* x          = (const float*)d_in[0];
  const int*   stage      = (const int*)d_in[1];
  const float* enc_w1     = (const float*)d_in[3];
  const float* enc_b1     = (const float*)d_in[4];
  const float* enc_w2     = (const float*)d_in[5];
  const float* enc_b2     = (const float*)d_in[6];
  const float* in_proj_w  = (const float*)d_in[7];
  const float* in_proj_b  = (const float*)d_in[8];
  const float* out_proj_w = (const float*)d_in[9];
  const float* out_proj_b = (const float*)d_in[10];
  const float* lstm_w_ih  = (const float*)d_in[11];
  const float* lstm_b_ih  = (const float*)d_in[13];
  const float* lstm_b_hh  = (const float*)d_in[14];
  const float* head_w     = (const float*)d_in[15];
  const float* head_b     = (const float*)d_in[16];

  char* w = (char*)d_ws;
  const size_t OFF_POS   = 0;                      // 5*512*4   = 10 KB
  const size_t OFF_WOV   = 0x8000;                 // 512*512*4 = 1 MB
  const size_t OFF_BAO   = OFF_WOV + 0x100000;     // 512*4
  const size_t OFF_DBIAS = OFF_BAO + 0x1000;       // 5*1536*4  = 30 KB
  const size_t OFF_MROW  = OFF_DBIAS + 0x8000;     // 1536*512*2 = 1.5 MB
  float*          posBuf = (float*)(w + OFF_POS);
  float*          Wov    = (float*)(w + OFF_WOV);
  float*          bao    = (float*)(w + OFF_BAO);
  float*          dbias  = (float*)(w + OFF_DBIAS);
  unsigned short* Mrow   = (unsigned short*)(w + OFF_MROW);

  float* out_feat   = (float*)d_out;
  float* out_logits = out_feat + (size_t)B_ROWS * DMODEL;

  pos_kernel        <<<10,   256, 0, stream>>>(enc_w1, enc_b1, enc_w2, enc_b2, posBuf);
  wov_kernel        <<<128,  256, 0, stream>>>(out_proj_w, in_proj_w, Wov);
  bao_kernel        <<<2,    256, 0, stream>>>(out_proj_w, out_proj_b, in_proj_b, bao);
  mrow_kernel       <<<384,  256, 0, stream>>>(lstm_w_ih, Wov, Mrow);
  dbias_kernel      <<<30,   256, 0, stream>>>(Mrow, posBuf, lstm_w_ih, bao,
                                               lstm_b_ih, lstm_b_hh, dbias);
  init_logits_kernel<<<(B_ROWS * 5 + 255) / 256, 256, 0, stream>>>(head_b, out_logits);
  fused_main_kernel <<<dim3(8, 512), 256, 0, stream>>>(x, stage, Mrow, dbias,
                                                       head_w, out_feat, out_logits);
}
